// QuantumHybridClassifier_65481071395749
// MI455X (gfx1250) — matrix operand order repaired and checked by reading, not compiled, not run on hardware
//
#include <hip/hip_runtime.h>

// ---------------------------------------------------------------------------
// QuantumHybridClassifier — CDNA5 (gfx1250)
//   conv1: implicit-GEMM WMMA (M=pixels,N=6->16,K=75->3x32) + fused pool, f16 out
//   conv2: implicit-GEMM WMMA (M=pixels,N=15->16,K=54->2x32) + fused pool -> A
//   FC1:   v_wmma_f32_16x16x32_f16, A staged via async global->LDS (double buf,
//          inline asm global_load_async_to_lds_b128 + s_wait_asynccnt),
//          K split 5-way deterministically, bias+relu reduce
//   tail:  exact fp32 GEMVs + sigmoid(-sin theta)
// ---------------------------------------------------------------------------

typedef __attribute__((ext_vector_type(16))) _Float16 v16h;
typedef __attribute__((ext_vector_type(8)))  _Float16 v8h;
typedef __attribute__((ext_vector_type(8)))  float    v8f;

#define BATCH 512
#define KREAL 55815
#define KPAD  56320               // 55815 padded to 440*128 (multiple of 128)
#define KSLICES 5
#define KCHUNK (KPAD / KSLICES)   // 11264 = 88 * 128
#define FC1_CHUNK 128             // halves staged per LDS chunk
#define FC1_NCHUNKS (KCHUNK / FC1_CHUNK)  // 88
#define AS_STRIDE 136             // padded LDS row stride (halves): conflict-free

#define FC1_ASYNC 1               // flip to 0 if async asm ever fails to assemble

// 16-byte global -> LDS async copy (ASYNCcnt path).
// vdst = 32-bit LDS byte offset (low 32 bits of a generic LDS pointer, per ISA
// aperture rule), vaddr = 64-bit global address, SADDR=off (GV mode).
static __device__ __forceinline__ void async_copy_b128(const void* gsrc, void* lds_dst) {
    const unsigned           lds = (unsigned)(size_t)lds_dst;
    const unsigned long long g   = (unsigned long long)(size_t)gsrc;
    asm volatile("global_load_async_to_lds_b128 %0, %1, off"
                 :: "v"(lds), "v"(g) : "memory");
}

static __device__ __forceinline__ void wait_asynccnt0() {
#if __has_builtin(__builtin_amdgcn_s_wait_asynccnt)
    __builtin_amdgcn_s_wait_asynccnt(0);
#else
    asm volatile("s_wait_asynccnt 0" ::: "memory");
#endif
}

// ===================================================================== conv1
// A-fragment gather from zero-padded LDS input tile.
// K mapping: k = ic*32 + (ky*5+kx), pad 25..31 per channel chunk.
static __device__ __forceinline__ v16h gatherA_c1(const _Float16* __restrict__ t,
                                                  int ch, int rowbase, int colbase,
                                                  int half) {
    union { v16h v; _Float16 e[16]; } a;
#pragma unroll
    for (int i = 0; i < 16; ++i) {
        const int kl = (i < 8) ? (8 * half + i) : (8 + 8 * half + i); // 16+8h+(i-8)
        const int ky = kl / 5, kx = kl % 5;
        _Float16 val = (_Float16)0.0f;
        if (kl < 25) val = t[(ch * 19 + rowbase + ky) * 36 + (colbase + kx)];
        a.e[i] = val;
    }
    return a.v;
}

__global__ __launch_bounds__(256)
void conv1_wmma_kernel(const float* __restrict__ in,
                       const float* __restrict__ w,
                       const float* __restrict__ bias,
                       _Float16* __restrict__ out) {
    __shared__ _Float16 tin[3 * 19 * 36];  // zero-padded f16 input tile
    __shared__ float swf[450];
    __shared__ float sbf[6];
    __shared__ float convt[6][8][16];      // relu'd conv outputs (8 rows x 16 cols)

    const int b   = blockIdx.y;
    const int pxb = blockIdx.x % 9, pyb = blockIdx.x / 9;
    const int py0 = pyb * 7, px0 = pxb * 15;
    const int tid = threadIdx.x;
    const int wave = tid >> 5, lane = tid & 31;
    const int half = lane >> 4, l16 = lane & 15;

    for (int i = tid; i < 450; i += 256) swf[i] = w[i];
    if (tid < 6) sbf[tid] = bias[tid];

    // stage input tile: rows 2*py0-1 .. +18, cols 2*px0-1 .. +34 (pad to 36), f32->f16
    const float* inb = in + (size_t)b * 187500;
    for (int i = tid; i < 3 * 19 * 36; i += 256) {
        const int cc = i % 36, t2 = i / 36, rr = t2 % 19, ic = t2 / 19;
        const int iy = 2 * py0 - 1 + rr, ix = 2 * px0 - 1 + cc;
        float v = 0.0f;
        if (iy >= 0 && iy < 250 && ix >= 0 && ix < 250)
            v = inb[(size_t)ic * 62500 + (size_t)iy * 250 + ix];
        tin[i] = (_Float16)v;
    }
    __syncthreads();

    // B fragments (weights), built once per lane: B[k][n], n = l16 (channel)
    v16h bfrag[3];
#pragma unroll
    for (int ch = 0; ch < 3; ++ch) {
        union { v16h v; _Float16 e[16]; } bb;
#pragma unroll
        for (int i = 0; i < 16; ++i) {
            const int kl = (i < 8) ? (8 * half + i) : (8 + 8 * half + i);
            float v = 0.0f;
            if (l16 < 6 && kl < 25) v = swf[(l16 * 3 + ch) * 25 + kl];
            bb.e[i] = (_Float16)v;
        }
        bfrag[ch] = bb.v;
    }

    // wave `wave` computes conv row ry = wave (16 cols), K accumulated over 3 chunks
    const int rowbase = 2 * wave;   // local iy base (iy_l = 2*ry + ky)
    const int colbase = 2 * l16;    // local ix base (ix_l = 2*cx + kx)
    v8f acc = {};
    acc = __builtin_amdgcn_wmma_f32_16x16x32_f16(false, gatherA_c1(tin, 0, rowbase, colbase, half),
                                                 false, bfrag[0], (short)0, acc, false, false);
    acc = __builtin_amdgcn_wmma_f32_16x16x32_f16(false, gatherA_c1(tin, 1, rowbase, colbase, half),
                                                 false, bfrag[1], (short)0, acc, false, false);
    acc = __builtin_amdgcn_wmma_f32_16x16x32_f16(false, gatherA_c1(tin, 2, rowbase, colbase, half),
                                                 false, bfrag[2], (short)0, acc, false, false);

    // C layout: lane -> N = l16 (channel), VGPR r -> M = r + 8*half (pixel col)
    if (l16 < 6) {
#pragma unroll
        for (int r = 0; r < 8; ++r)
            convt[l16][wave][r + 8 * half] = fmaxf(acc[r] + sbf[l16], 0.0f);
    }
    __syncthreads();

    // fused maxpool 2x2 stride1, write f16 pooled1
    const int prows = (123 - py0 < 7) ? (123 - py0) : 7;
    const int pcols = (123 - px0 < 15) ? (123 - px0) : 15;
    const int ptotal = 6 * prows * pcols;
    for (int i = tid; i < ptotal; i += 256) {
        const int pxl = i % pcols;
        const int ryp = (i / pcols) % prows;
        const int c   = i / (pcols * prows);
        const float m = fmaxf(fmaxf(convt[c][ryp][pxl],     convt[c][ryp][pxl + 1]),
                              fmaxf(convt[c][ryp + 1][pxl], convt[c][ryp + 1][pxl + 1]));
        out[((size_t)b * 6 + c) * (123 * 123) + (size_t)(py0 + ryp) * 123 + (px0 + pxl)] =
            (_Float16)m;
    }
}

// ===================================================================== conv2
// K mapping: k = ic*9 + ky*3 + kx (0..53), pad 54..63.
static __device__ __forceinline__ v16h gatherA_c2(const _Float16* __restrict__ t,
                                                  int chunk, int rowbase, int colbase,
                                                  int half) {
    union { v16h v; _Float16 e[16]; } a;
#pragma unroll
    for (int i = 0; i < 16; ++i) {
        const int kl = (i < 8) ? (8 * half + i) : (8 + 8 * half + i);
        const int k  = 32 * chunk + kl;
        const int ic = k / 9, j = k % 9, ky = j / 3, kx = j % 3;
        _Float16 val = (_Float16)0.0f;
        if (k < 54) val = t[(ic * 17 + rowbase + ky) * 36 + (colbase + kx)];
        a.e[i] = val;
    }
    return a.v;
}

__global__ __launch_bounds__(256)
void conv2_wmma_kernel(const _Float16* __restrict__ in,
                       const float* __restrict__ w,
                       const float* __restrict__ bias,
                       _Float16* __restrict__ A) {
    __shared__ _Float16 tin[6 * 17 * 36];
    __shared__ float swf[810];
    __shared__ float sbf[15];
    __shared__ float convt[15][8][16];

    const int b   = blockIdx.y;
    const int pxb = blockIdx.x % 5, pyb = blockIdx.x / 5;
    const int py0 = pyb * 7, px0 = pxb * 15;
    const int tid = threadIdx.x;
    const int wave = tid >> 5, lane = tid & 31;
    const int half = lane >> 4, l16 = lane & 15;

    for (int i = tid; i < 810; i += 256) swf[i] = w[i];
    if (tid < 15) sbf[tid] = bias[tid];

    // stage pooled1 tile: rows 2*py0-1 .. +16, cols 2*px0-1 .. +34 (pad 36)
    for (int i = tid; i < 6 * 17 * 36; i += 256) {
        const int cc = i % 36, t2 = i / 36, rr = t2 % 17, ic = t2 / 17;
        const int iy = 2 * py0 - 1 + rr, ix = 2 * px0 - 1 + cc;
        _Float16 v = (_Float16)0.0f;
        if (iy >= 0 && iy < 123 && ix >= 0 && ix < 123)
            v = in[((size_t)b * 6 + ic) * (123 * 123) + (size_t)iy * 123 + ix];
        tin[i] = v;
    }
    __syncthreads();

    // B fragments: B[k][n], n = l16 (out channel), w2 flat = n*54 + k
    v16h bfrag[2];
#pragma unroll
    for (int ch = 0; ch < 2; ++ch) {
        union { v16h v; _Float16 e[16]; } bb;
#pragma unroll
        for (int i = 0; i < 16; ++i) {
            const int kg = 32 * ch + ((i < 8) ? (8 * half + i) : (8 + 8 * half + i));
            float v = 0.0f;
            if (l16 < 15 && kg < 54) v = swf[l16 * 54 + kg];
            bb.e[i] = (_Float16)v;
        }
        bfrag[ch] = bb.v;
    }

    const int rowbase = 2 * wave;
    const int colbase = 2 * l16;
    v8f acc = {};
    acc = __builtin_amdgcn_wmma_f32_16x16x32_f16(false, gatherA_c2(tin, 0, rowbase, colbase, half),
                                                 false, bfrag[0], (short)0, acc, false, false);
    acc = __builtin_amdgcn_wmma_f32_16x16x32_f16(false, gatherA_c2(tin, 1, rowbase, colbase, half),
                                                 false, bfrag[1], (short)0, acc, false, false);

    if (l16 < 15) {
#pragma unroll
        for (int r = 0; r < 8; ++r)
            convt[l16][wave][r + 8 * half] = fmaxf(acc[r] + sbf[l16], 0.0f);
    }
    __syncthreads();

    // fused maxpool, write f16 directly into padded flattened FC1 A layout
    const int prows = (61 - py0 < 7) ? (61 - py0) : 7;
    const int pcols = (61 - px0 < 15) ? (61 - px0) : 15;
    const int ptotal = 15 * prows * pcols;
    for (int i = tid; i < ptotal; i += 256) {
        const int pxl = i % pcols;
        const int ryp = (i / pcols) % prows;
        const int c   = i / (pcols * prows);
        const float m = fmaxf(fmaxf(convt[c][ryp][pxl],     convt[c][ryp][pxl + 1]),
                              fmaxf(convt[c][ryp + 1][pxl], convt[c][ryp + 1][pxl + 1]));
        A[(size_t)b * KPAD + (size_t)c * 3721 + (size_t)(py0 + ryp) * 61 + (px0 + pxl)] =
            (_Float16)m;
    }
    // zero the K padding once per image
    if (blockIdx.x == 0) {
        for (int i = tid; i < KPAD - KREAL; i += 256)
            A[(size_t)b * KPAD + KREAL + i] = (_Float16)0.0f;
    }
}

// --------------------------------------------------- fc1 weight fp32 -> f16
__global__ __launch_bounds__(256)
void wcvt_kernel(const float* __restrict__ w, _Float16* __restrict__ wh) {
    size_t idx = (size_t)blockIdx.x * 256 + threadIdx.x;
    if (idx >= (size_t)128 * KPAD) return;
    const int n = (int)(idx / KPAD);
    const int k = (int)(idx % KPAD);
    float v = (n < 120 && k < KREAL) ? w[(size_t)n * KREAL + k] : 0.0f;
    wh[idx] = (_Float16)v;
}

// ------------------------------------------------------------- FC1 via WMMA
// grid = (KSLICES, 32 M-tiles), block 256 (8 waves). Wave w owns N-tile w.
// A tile (16 rows, shared by all waves) is staged global->LDS asynchronously.
__global__ __launch_bounds__(256)
void fc1_wmma_kernel(const _Float16* __restrict__ A,
                     const _Float16* __restrict__ Wh,
                     float* __restrict__ part) {
    const int tid  = threadIdx.x;
    const int wave = tid >> 5;
    const int lane = tid & 31;
    const int ks   = blockIdx.x;
    const int mt   = blockIdx.y;
    const int half = lane >> 4;
    const int l16  = lane & 15;

    const _Float16* brow = Wh + (size_t)(wave * 16 + l16) * KPAD + (size_t)ks * KCHUNK;
    v8f c = {};

#if FC1_ASYNC
    __shared__ _Float16 As[2][16][AS_STRIDE];   // double-buffered A chunk
    const int srow = tid >> 4;                  // 0..15 : A row
    const int sseg = tid & 15;                  // 0..15 : 8-half (16B) segment
    const _Float16* gsrc = A + (size_t)(mt * 16 + srow) * KPAD
                             + (size_t)ks * KCHUNK + sseg * 8;
    // prologue: stage chunk 0
    async_copy_b128(gsrc, &As[0][srow][sseg * 8]);
    wait_asynccnt0();
    __syncthreads();

    int buf = 0;
    for (int ch = 0; ch < FC1_NCHUNKS; ++ch) {
        if (ch + 1 < FC1_NCHUNKS) {
            async_copy_b128(gsrc + (size_t)(ch + 1) * FC1_CHUNK,
                            &As[buf ^ 1][srow][sseg * 8]);
        }
        const _Float16* arow = &As[buf][l16][0];
        const _Float16* bchk = brow + (size_t)ch * FC1_CHUNK;
#pragma unroll
        for (int kk = 0; kk < FC1_CHUNK; kk += 32) {
            union { v16h v; v8h h[2]; } a, bm;
            a.h[0]  = *(const v8h*)(arow + kk + 8 * half);
            a.h[1]  = *(const v8h*)(arow + kk + 16 + 8 * half);
            bm.h[0] = *(const v8h*)(bchk + kk + 8 * half);
            bm.h[1] = *(const v8h*)(bchk + kk + 16 + 8 * half);
            c = __builtin_amdgcn_wmma_f32_16x16x32_f16(false, a.v, false, bm.v,
                                                       (short)0, c, false, false);
        }
        wait_asynccnt0();      // own async stores to LDS done
        __syncthreads();       // all waves' stores visible
        buf ^= 1;
    }
#else
    const _Float16* arow = A + (size_t)(mt * 16 + l16) * KPAD + (size_t)ks * KCHUNK;
    for (int k0 = 0; k0 < KCHUNK; k0 += 32) {
        union { v16h v; v8h h[2]; } a, bm;
        a.h[0]  = *(const v8h*)(arow + k0 + 8 * half);
        a.h[1]  = *(const v8h*)(arow + k0 + 16 + 8 * half);
        bm.h[0] = *(const v8h*)(brow + k0 + 16 * half);
        bm.h[1] = *(const v8h*)(brow + k0 + 16 * half + 8);
        __builtin_prefetch(arow + k0 + 64, 0, 1);
        __builtin_prefetch(brow + k0 + 64, 0, 1);
        c = __builtin_amdgcn_wmma_f32_16x16x32_f16(false, a.v, false, bm.v,
                                                   (short)0, c, false, false);
    }
#endif

    // C layout: VGPR r, lane -> M = r + 8*half, N = l16
#pragma unroll
    for (int r = 0; r < 8; ++r) {
        const int m = mt * 16 + r + 8 * half;
        part[((size_t)ks * BATCH + m) * 128 + wave * 16 + l16] = c[r];
    }
}

__global__ __launch_bounds__(256)
void fc1_finish_kernel(const float* __restrict__ part,
                       const float* __restrict__ bias,
                       float* __restrict__ x1) {
    const int idx = blockIdx.x * 256 + threadIdx.x;
    if (idx >= BATCH * 120) return;
    const int b = idx / 120, n = idx % 120;
    float acc = bias[n];
    for (int s = 0; s < KSLICES; ++s)
        acc += part[((size_t)s * BATCH + b) * 128 + n];
    x1[idx] = fmaxf(acc, 0.0f);
}

// ------------------------------------------------------------------- tail
__global__ __launch_bounds__(256)
void gemv120_kernel(const float* __restrict__ x, const float* __restrict__ W,
                    const float* __restrict__ bvec, float* __restrict__ out,
                    int nout, int relu) {
    const int idx = blockIdx.x * 256 + threadIdx.x;
    if (idx >= BATCH * nout) return;
    const int b = idx / nout, n = idx % nout;
    const float* xr = x + (size_t)b * 120;
    const float* wr = W + (size_t)n * 120;
    float acc = bvec[n];
    for (int j = 0; j < 120; ++j) acc = fmaf(xr[j], wr[j], acc);
    out[idx] = relu ? fmaxf(acc, 0.0f) : acc;
}

__global__ __launch_bounds__(256)
void head_kernel(const float* __restrict__ z, const float* __restrict__ fc3_w,
                 const float* __restrict__ fc3_b, float* __restrict__ out) {
    const int b = blockIdx.x * 256 + threadIdx.x;
    if (b >= BATCH) return;
    const float* zr = z + (size_t)b * 84;
    float acc = fc3_b[0];
    for (int j = 0; j < 84; ++j) acc = fmaf(zr[j], fc3_w[j], acc);
    const float s = 1.0f / (1.0f + expf(sinf(acc)));  // sigmoid(-sin theta)
    out[b] = s;
    out[b + BATCH] = 1.0f - s;
}

// ---------------------------------------------------------------------------
extern "C" void kernel_launch(void* const* d_in, const int* in_sizes, int n_in,
                              void* d_out, int out_size, void* d_ws, size_t ws_size,
                              hipStream_t stream) {
    const float* inputs     = (const float*)d_in[0];
    const float* conv1_w    = (const float*)d_in[1];
    const float* conv1_b    = (const float*)d_in[2];
    const float* conv2_w    = (const float*)d_in[3];
    const float* conv2_b    = (const float*)d_in[4];
    const float* fc1_w      = (const float*)d_in[5];
    const float* fc1_b      = (const float*)d_in[6];
    const float* in_proj_w  = (const float*)d_in[7];
    const float* in_proj_b  = (const float*)d_in[8];
    const float* out_proj_w = (const float*)d_in[9];
    const float* out_proj_b = (const float*)d_in[10];
    const float* fc2_w      = (const float*)d_in[11];
    const float* fc2_b      = (const float*)d_in[12];
    const float* fc3_w      = (const float*)d_in[13];
    const float* fc3_b      = (const float*)d_in[14];
    float* out = (float*)d_out;

    char* ws = (char*)d_ws;
    size_t off = 0;
    auto wsalloc = [&](size_t bytes) -> void* {
        void* p = ws + off;
        off = (off + bytes + 255) & ~(size_t)255;
        return p;
    };
    _Float16* pooled1 = (_Float16*)wsalloc((size_t)BATCH * 6 * 123 * 123 * sizeof(_Float16));
    _Float16* Abuf    = (_Float16*)wsalloc((size_t)BATCH * KPAD * sizeof(_Float16));
    _Float16* Wh      = (_Float16*)wsalloc((size_t)128 * KPAD * sizeof(_Float16));
    float*    part    = (float*)wsalloc((size_t)KSLICES * BATCH * 128 * sizeof(float));
    float*    x1      = (float*)wsalloc((size_t)BATCH * 120 * sizeof(float));
    float*    vbuf    = (float*)wsalloc((size_t)BATCH * 120 * sizeof(float));
    float*    ybuf    = (float*)wsalloc((size_t)BATCH * 120 * sizeof(float));
    float*    zbuf    = (float*)wsalloc((size_t)BATCH * 84 * sizeof(float));
    (void)ws_size; (void)in_sizes; (void)n_in; (void)out_size;

    // 1) conv1 + relu + pool via WMMA implicit GEMM (9 col-blocks x 18 row-blocks)
    conv1_wmma_kernel<<<dim3(9 * 18, BATCH), 256, 0, stream>>>(
        inputs, conv1_w, conv1_b, pooled1);

    // 2) conv2 + relu + pool via WMMA implicit GEMM (5 x 9 blocks) -> padded A
    conv2_wmma_kernel<<<dim3(5 * 9, BATCH), 256, 0, stream>>>(
        pooled1, conv2_w, conv2_b, Abuf);

    // 3) fc1 weight fp32 -> f16 padded [128, KPAD]
    {
        const size_t total = (size_t)128 * KPAD;
        wcvt_kernel<<<(unsigned)((total + 255) / 256), 256, 0, stream>>>(fc1_w, Wh);
    }

    // 4) FC1 GEMM via WMMA (async LDS-staged A), deterministic K-split reduce
    fc1_wmma_kernel<<<dim3(KSLICES, BATCH / 16), 256, 0, stream>>>(Abuf, Wh, part);
    fc1_finish_kernel<<<(BATCH * 120 + 255) / 256, 256, 0, stream>>>(part, fc1_b, x1);

    // 5) attention (seq_len==1): v = x @ Wv^T + bv ; y = v @ Wo^T + bo
    gemv120_kernel<<<(BATCH * 120 + 255) / 256, 256, 0, stream>>>(
        x1, in_proj_w + 240 * 120, in_proj_b + 240, vbuf, 120, 0);
    gemv120_kernel<<<(BATCH * 120 + 255) / 256, 256, 0, stream>>>(
        vbuf, out_proj_w, out_proj_b, ybuf, 120, 0);

    // 6) fc2 + relu
    gemv120_kernel<<<(BATCH * 84 + 255) / 256, 256, 0, stream>>>(
        ybuf, fc2_w, fc2_b, zbuf, 84, 1);

    // 7) fc3 + quantum head + sigmoid, concat [p, 1-p]
    head_kernel<<<(BATCH + 255) / 256, 256, 0, stream>>>(zbuf, fc3_w, fc3_b, out);
}
